// MambaLikeSSM_82549271429744
// MI455X (gfx1250) — matrix operand order repaired and checked by reading, not compiled, not run on hardware
//
#include <hip/hip_runtime.h>
#include <math.h>
#include <stdint.h>

typedef __attribute__((ext_vector_type(16))) __bf16       v16bf;
typedef __attribute__((ext_vector_type(8)))  __bf16       v8bf;
typedef __attribute__((ext_vector_type(8)))  float        v8f;
typedef __attribute__((ext_vector_type(4)))  unsigned int u32x4;

union FragU { v16bf v; u32x4 q[2]; };

// ---------------- async global->LDS copy (CDNA5), with safe fallback ----------------
#if defined(__has_builtin)
#if __has_builtin(__builtin_amdgcn_global_load_async_to_lds_b128)
#define HAVE_ASYNC_CP 1
#endif
#if __has_builtin(__builtin_amdgcn_s_wait_asynccnt)
#define HAVE_ASYNC_WAIT 1
#endif
#endif

// builtin expects pointer-to-int-__vector(4)
typedef int v4i_gcc __attribute__((vector_size(16)));
typedef __attribute__((address_space(1))) v4i_gcc as1_v4i;
typedef __attribute__((address_space(3))) v4i_gcc as3_v4i;

__device__ __forceinline__ void cp16(void* l, const void* g) {
#if defined(HAVE_ASYNC_CP)
  __builtin_amdgcn_global_load_async_to_lds_b128(
      (as1_v4i*)(uintptr_t)g,
      (as3_v4i*)(uintptr_t)(unsigned)(uintptr_t)l, 0, 0);
#else
  *(u32x4*)l = *(const u32x4*)g;
#endif
}

__device__ __forceinline__ void wait_cp() {
#if defined(HAVE_ASYNC_CP)
#if defined(HAVE_ASYNC_WAIT)
  __builtin_amdgcn_s_wait_asynccnt(0);
#else
  asm volatile("s_wait_asynccnt 0x0" ::: "memory");
#endif
#endif
}

__device__ __forceinline__ v8f zero8() {
  v8f z;
#pragma unroll
  for (int i = 0; i < 8; ++i) z[i] = 0.0f;
  return z;
}

__device__ __forceinline__ float sigm(float x) { return 1.0f / (1.0f + __expf(-x)); }

// LDS-resident fragment loaders (row pointer already includes row base; tile K = 32).
__device__ __forceinline__ v16bf fragA_lds(const __bf16* row, int hi) {
  const __bf16* p = row + 8 * hi;
  FragU f;
  f.q[0] = *(const u32x4*)(p);
  f.q[1] = *(const u32x4*)(p + 16);
  return f.v;
}
__device__ __forceinline__ v16bf fragB_lds(const __bf16* row, int hi) {
  const __bf16* p = row + 8 * hi;
  FragU f;
  f.q[0] = *(const u32x4*)(p);
  f.q[1] = *(const u32x4*)(p + 16);
  return f.v;
}

#define WMMA_BF16(a, b, c) \
  __builtin_amdgcn_wmma_f32_16x16x32_bf16(false, (a), false, (b), (short)0, (c), false, false)

#define KT 32
#define LDP 40  // padded LDS row (elements): 80B rows, 16B aligned, bank-spread

// ----------------------------------------------------------------
// Shared GEMM phase: workgroup tile 64M x 128N, LDS double-buffered,
// async copies for iteration k+32 issued before the WMMAs of iteration k.
// DUAL=true accumulates two B matrices against the same A fragments.
template <bool DUAL>
__device__ __forceinline__ void tile_phase(__bf16 (&As)[2][64][LDP],
                                           __bf16 (&Bs)[2][2][128][LDP],
                                           const __bf16* __restrict__ Ag, int lda,
                                           const __bf16* __restrict__ Bg0,
                                           const __bf16* __restrict__ Bg1, int ldb, int K,
                                           int M0, int N0, int tid, v8f acc0[4], v8f acc1[4]) {
  int lane = tid & 31;
  int hi = lane >> 4, ln = lane & 15;
  int w = tid >> 5;
  int mSub = w >> 1, nSub = w & 1;
  int crow = tid >> 2;       // 0..63
  int cseg = (tid & 3) * 8;  // element offset within 32-wide K slice
  int mLoc = mSub * 16 + ln;

  // prologue: stage k = 0
  cp16(&As[0][crow][cseg], Ag + (size_t)(M0 + crow) * lda + cseg);
  cp16(&Bs[0][0][crow][cseg], Bg0 + (size_t)(N0 + crow) * ldb + cseg);
  cp16(&Bs[0][0][crow + 64][cseg], Bg0 + (size_t)(N0 + crow + 64) * ldb + cseg);
  if constexpr (DUAL) {
    cp16(&Bs[0][1][crow][cseg], Bg1 + (size_t)(N0 + crow) * ldb + cseg);
    cp16(&Bs[0][1][crow + 64][cseg], Bg1 + (size_t)(N0 + crow + 64) * ldb + cseg);
  }
  wait_cp();
  __syncthreads();

  int buf = 0;
  for (int k = 0; k < K; k += KT) {
    int nk = k + KT;
    if (nk < K) {
      cp16(&As[buf ^ 1][crow][cseg], Ag + (size_t)(M0 + crow) * lda + nk + cseg);
      cp16(&Bs[buf ^ 1][0][crow][cseg], Bg0 + (size_t)(N0 + crow) * ldb + nk + cseg);
      cp16(&Bs[buf ^ 1][0][crow + 64][cseg], Bg0 + (size_t)(N0 + crow + 64) * ldb + nk + cseg);
      if constexpr (DUAL) {
        cp16(&Bs[buf ^ 1][1][crow][cseg], Bg1 + (size_t)(N0 + crow) * ldb + nk + cseg);
        cp16(&Bs[buf ^ 1][1][crow + 64][cseg], Bg1 + (size_t)(N0 + crow + 64) * ldb + nk + cseg);
      }
    }
    v16bf a = fragA_lds(&As[buf][mLoc][0], hi);
#pragma unroll
    for (int j = 0; j < 4; ++j) {
      int nl = nSub * 64 + j * 16 + ln;
      acc0[j] = WMMA_BF16(a, fragB_lds(&Bs[buf][0][nl][0], hi), acc0[j]);
      if constexpr (DUAL) acc1[j] = WMMA_BF16(a, fragB_lds(&Bs[buf][1][nl][0], hi), acc1[j]);
    }
    wait_cp();
    __syncthreads();
    buf ^= 1;
  }
}

// ---------------------------------------------------------------- convert f32 -> bf16 (8/thread)
__global__ void cvt_f32_bf16_v8(const float* __restrict__ in, __bf16* __restrict__ out, int n8) {
  int i = blockIdx.x * blockDim.x + threadIdx.x;
  int stride = gridDim.x * blockDim.x;
  for (; i < n8; i += stride) {
    const float4* p = (const float4*)in + (size_t)i * 2;
    float4 f0 = p[0], f1 = p[1];
    v8bf o;
    o[0] = (__bf16)f0.x; o[1] = (__bf16)f0.y; o[2] = (__bf16)f0.z; o[3] = (__bf16)f0.w;
    o[4] = (__bf16)f1.x; o[5] = (__bf16)f1.y; o[6] = (__bf16)f1.z; o[7] = (__bf16)f1.w;
    *(v8bf*)(out + (size_t)i * 8) = o;
  }
}

// ---------------------------------------------------------------- GEMM1: H1 = silu(X @ W1^T + b1)
// grid: (M/64) * (512/128) blocks
__global__ __launch_bounds__(256) void gemm_silu_kernel(const __bf16* __restrict__ X,
                                                        const __bf16* __restrict__ W,
                                                        const float* __restrict__ bias,
                                                        __bf16* __restrict__ Out) {
  __shared__ __bf16 As[2][64][LDP];
  __shared__ __bf16 Bs[2][2][128][LDP];

  int tid = threadIdx.x;
  int lane = tid & 31, hi = lane >> 4, ln = lane & 15;
  int w = tid >> 5, mSub = w >> 1, nSub = w & 1;
  int M0 = (blockIdx.x >> 2) * 64, N0 = (blockIdx.x & 3) * 128;

  v8f acc[4];
#pragma unroll
  for (int j = 0; j < 4; ++j) acc[j] = zero8();

  tile_phase<false>(As, Bs, X, 1024, W, nullptr, 1024, 1024, M0, N0, tid, acc, acc);

#pragma unroll
  for (int j = 0; j < 4; ++j) {
    int n = N0 + nSub * 64 + j * 16 + ln;
    float bn = bias[n];
#pragma unroll
    for (int v = 0; v < 8; ++v) {
      int mm = M0 + mSub * 16 + v + 8 * hi;
      float val = acc[j][v] + bn;
      Out[(size_t)mm * 512 + n] = (__bf16)(val * sigm(val));
    }
  }
}

// ---------------------------------------------------------------- u = (X@Bw^T + Bb) * sigmoid(H1@W2^T + sb2)
// grid: M/64 blocks (N = 128 fits one tile column). Two K-phases with different A sources.
__global__ __launch_bounds__(256) void proj_u_kernel(const __bf16* __restrict__ Xb,
                                                     const __bf16* __restrict__ Bw,
                                                     const float* __restrict__ Bb,
                                                     const __bf16* __restrict__ H1,
                                                     const __bf16* __restrict__ W2,
                                                     const float* __restrict__ sb2,
                                                     float* __restrict__ u) {
  __shared__ __bf16 As[2][64][LDP];
  __shared__ __bf16 Bs[2][2][128][LDP];

  int tid = threadIdx.x;
  int lane = tid & 31, hi = lane >> 4, ln = lane & 15;
  int w = tid >> 5, mSub = w >> 1, nSub = w & 1;
  int M0 = blockIdx.x * 64;

  v8f aB[4], aS[4];
#pragma unroll
  for (int j = 0; j < 4; ++j) { aB[j] = zero8(); aS[j] = zero8(); }

  tile_phase<false>(As, Bs, Xb, 1024, Bw, nullptr, 1024, 1024, M0, 0, tid, aB, aB);
  tile_phase<false>(As, Bs, H1, 512, W2, nullptr, 512, 512, M0, 0, tid, aS, aS);

#pragma unroll
  for (int j = 0; j < 4; ++j) {
    int n = nSub * 64 + j * 16 + ln;
    float bb = Bb[n], s2 = sb2[n];
#pragma unroll
    for (int v = 0; v < 8; ++v) {
      int mm = M0 + mSub * 16 + v + 8 * hi;
      u[(size_t)mm * 128 + n] = (aB[j][v] + bb) * sigm(aS[j][v] + s2);
    }
  }
}

// ---------------------------------------------------------------- P squaring: D = S @ S (128x128 f32)
__global__ void matsq_kernel(const float* __restrict__ S, float* __restrict__ D) {
  int o = blockIdx.x * blockDim.x + threadIdx.x;
  int r = o >> 7, c = o & 127;
  float s = 0.0f;
  for (int k = 0; k < 128; ++k) s += S[r * 128 + k] * S[k * 128 + c];
  D[o] = s;
}

// ---------------------------------------------------------------- chunked recurrence (f32)
__global__ __launch_bounds__(256) void chunk_kernel(const float* __restrict__ A,
                                                    const float* __restrict__ u,
                                                    const float* __restrict__ E,
                                                    float* __restrict__ F,
                                                    __bf16* __restrict__ hs, int pass) {
  __shared__ float hbuf[2][128];
  __shared__ float psum[128];
  int t = threadIdx.x;
  int half = t >> 7, i = t & 127;
  int g = blockIdx.x;
  int b = g >> 4, c = g & 15;

  float ar[64];
#pragma unroll
  for (int jj = 0; jj < 64; ++jj) ar[jj] = A[i * 128 + half * 64 + jj];

  if (half == 0) hbuf[0][i] = (pass == 2) ? E[g * 128 + i] : 0.0f;
  __syncthreads();

  const float* ub = u + ((size_t)(b * 2048 + c * 128)) * 128;
  int cur = 0;
  for (int tl = 0; tl < 128; ++tl) {
    float p = 0.0f;
#pragma unroll
    for (int jj = 0; jj < 64; ++jj) p += ar[jj] * hbuf[cur][half * 64 + jj];
    if (half) psum[i] = p;
    __syncthreads();
    if (!half) {
      float hn = p + psum[i] + ub[tl * 128 + i];
      hbuf[cur ^ 1][i] = hn;
      if (pass == 2) hs[((size_t)(b * 2048 + c * 128 + tl)) * 128 + i] = (__bf16)hn;
    }
    __syncthreads();
    cur ^= 1;
  }
  if (pass == 1 && half == 0) F[g * 128 + i] = hbuf[cur][i];
}

// ---------------------------------------------------------------- serial chain: E_c = P E_{c-1} + F_{c-1}
__global__ __launch_bounds__(1024) void chain_kernel(const float* __restrict__ P,
                                                     const float* __restrict__ F,
                                                     float* __restrict__ E) {
  __shared__ float eb[8][128];
  int t = threadIdx.x;
  int b = t >> 7, i = t & 127;
  eb[b][i] = 0.0f;
  E[(b * 16 + 0) * 128 + i] = 0.0f;
  __syncthreads();
  for (int c = 1; c < 16; ++c) {
    float s = F[(b * 16 + c - 1) * 128 + i];
    for (int k = 0; k < 128; ++k) s += P[i * 128 + k] * eb[b][k];
    __syncthreads();
    eb[b][i] = s;
    E[(b * 16 + c) * 128 + i] = s;
    __syncthreads();
  }
}

// ---------------------------------------------------------------- fused output:
// out = (hs@Cw^T + Cb + X@Dw^T + Db) * sigmoid(X@Gw^T + gb)
// grid: (M/64) * (1024/128) blocks
__global__ __launch_bounds__(256) void out_kernel(const __bf16* __restrict__ hsb,
                                                  const __bf16* __restrict__ Cw,
                                                  const float* __restrict__ Cb,
                                                  const __bf16* __restrict__ Xb,
                                                  const __bf16* __restrict__ Dw,
                                                  const float* __restrict__ Db,
                                                  const __bf16* __restrict__ Gw,
                                                  const float* __restrict__ gb,
                                                  float* __restrict__ out) {
  __shared__ __bf16 As[2][64][LDP];
  __shared__ __bf16 Bs[2][2][128][LDP];

  int tid = threadIdx.x;
  int lane = tid & 31, hi = lane >> 4, ln = lane & 15;
  int w = tid >> 5, mSub = w >> 1, nSub = w & 1;
  int M0 = (blockIdx.x >> 3) * 64, N0 = (blockIdx.x & 7) * 128;

  v8f accY[4], accG[4];
#pragma unroll
  for (int j = 0; j < 4; ++j) { accY[j] = zero8(); accG[j] = zero8(); }

  // phase 1: hs @ Cw^T (K=128); phase 2: X @ {Dw,Gw}^T (K=1024) sharing A fragments
  tile_phase<false>(As, Bs, hsb, 128, Cw, nullptr, 128, 128, M0, N0, tid, accY, accY);
  tile_phase<true>(As, Bs, Xb, 1024, Dw, Gw, 1024, 1024, M0, N0, tid, accY, accG);

#pragma unroll
  for (int j = 0; j < 4; ++j) {
    int n = N0 + nSub * 64 + j * 16 + ln;
    float cy = Cb[n] + Db[n];
    float cg = gb[n];
#pragma unroll
    for (int v = 0; v < 8; ++v) {
      int mm = M0 + mSub * 16 + v + 8 * hi;
      out[(size_t)mm * 1024 + n] = (accY[j][v] + cy) * sigm(accG[j][v] + cg);
    }
  }
}

// ================================================================ host
extern "C" void kernel_launch(void* const* d_in, const int* in_sizes, int n_in, void* d_out,
                              int out_size, void* d_ws, size_t ws_size, hipStream_t stream) {
  const float* x    = (const float*)d_in[0];
  const float* A    = (const float*)d_in[1];
  const float* B_w  = (const float*)d_in[2];
  const float* B_b  = (const float*)d_in[3];
  const float* C_w  = (const float*)d_in[4];
  const float* C_b  = (const float*)d_in[5];
  const float* D_w  = (const float*)d_in[6];
  const float* D_b  = (const float*)d_in[7];
  const float* s_w1 = (const float*)d_in[8];
  const float* s_b1 = (const float*)d_in[9];
  const float* s_w2 = (const float*)d_in[10];
  const float* s_b2 = (const float*)d_in[11];
  const float* g_w  = (const float*)d_in[12];
  const float* g_b  = (const float*)d_in[13];
  float* out = (float*)d_out;

  const int M = 16384;  // 8 * 2048

  char* w = (char*)d_ws;
  auto carve = [&](size_t bytes) -> char* {
    char* p = w;
    w += (bytes + 255) & ~(size_t)255;
    return p;
  };
  __bf16* Xb  = (__bf16*)carve((size_t)M * 1024 * 2);
  __bf16* W1b = (__bf16*)carve((size_t)512 * 1024 * 2);
  __bf16* W2b = (__bf16*)carve((size_t)128 * 512 * 2);
  __bf16* Bwb = (__bf16*)carve((size_t)128 * 1024 * 2);
  __bf16* Cwb = (__bf16*)carve((size_t)1024 * 128 * 2);
  __bf16* Dwb = (__bf16*)carve((size_t)1024 * 1024 * 2);
  __bf16* Gwb = (__bf16*)carve((size_t)1024 * 1024 * 2);
  __bf16* H1b = (__bf16*)carve((size_t)M * 512 * 2);
  float*  u   = (float*)carve((size_t)M * 128 * 4);
  __bf16* hsb = (__bf16*)carve((size_t)M * 128 * 2);
  float*  F   = (float*)carve((size_t)128 * 128 * 4);
  float*  E   = (float*)carve((size_t)128 * 128 * 4);
  float*  P0  = (float*)carve((size_t)128 * 128 * 4);
  float*  P1  = (float*)carve((size_t)128 * 128 * 4);

  // 1) precision conversion (8 elements / thread)
  cvt_f32_bf16_v8<<<2048, 256, 0, stream>>>(x, Xb, M * 1024 / 8);
  cvt_f32_bf16_v8<<<256, 256, 0, stream>>>(s_w1, W1b, 512 * 1024 / 8);
  cvt_f32_bf16_v8<<<32, 256, 0, stream>>>(s_w2, W2b, 128 * 512 / 8);
  cvt_f32_bf16_v8<<<64, 256, 0, stream>>>(B_w, Bwb, 128 * 1024 / 8);
  cvt_f32_bf16_v8<<<64, 256, 0, stream>>>(C_w, Cwb, 1024 * 128 / 8);
  cvt_f32_bf16_v8<<<512, 256, 0, stream>>>(D_w, Dwb, 1024 * 1024 / 8);
  cvt_f32_bf16_v8<<<512, 256, 0, stream>>>(g_w, Gwb, 1024 * 1024 / 8);

  // 2) H1 = silu(X @ s_w1^T + s_b1)
  gemm_silu_kernel<<<(M / 64) * (512 / 128), 256, 0, stream>>>(Xb, W1b, s_b1, H1b);

  // 3) u = (X@B_w^T + B_b) * sigmoid(H1@s_w2^T + s_b2)
  proj_u_kernel<<<M / 64, 256, 0, stream>>>(Xb, Bwb, B_b, H1b, W2b, s_b2, u);

  // 4) P = A^128 via 7 squarings (ping-pong); result lands in P0
  matsq_kernel<<<64, 256, 0, stream>>>(A, P0);
  const float* src = P0;
  float* dst = P1;
  for (int it = 1; it < 7; ++it) {
    matsq_kernel<<<64, 256, 0, stream>>>(src, dst);
    const float* tmp = dst; dst = (float*)src; src = tmp;
  }

  // 5) chunk pass 1 -> F ;  6) serial chain -> E ;  7) chunk pass 2 -> hs (bf16)
  chunk_kernel<<<128, 256, 0, stream>>>(A, u, E, F, hsb, 1);
  chain_kernel<<<1, 1024, 0, stream>>>(src, F, E);
  chunk_kernel<<<128, 256, 0, stream>>>(A, u, E, F, hsb, 2);

  // 8) fused output GEMMs + gate
  out_kernel<<<(M / 64) * (1024 / 128), 256, 0, stream>>>(hsb, Cwb, C_b, Xb, Dwb, D_b, Gwb, g_b,
                                                          out);
  (void)in_sizes; (void)n_in; (void)out_size; (void)ws_size;
}
